// GCNLayer_30270929502676
// MI455X (gfx1250) — matrix operand order repaired and checked by reading, not compiled, not run on hardware
//
#include <hip/hip_runtime.h>

// ---------------------------------------------------------------------------
// GCN layer for MI455X (gfx1250, wave32):
//   dis = rsqrt(rowsum(A) + 1 + 1e-10)
//   Y   = dis[:,None] * (X @ W)           (fp32 WMMA 16x16x4, exact)
//   out = relu( dis[:,None] * (A @ Y + Y) ) (bf16 WMMA 16x16x32, fp32 acc)
// N=8192, F=512, C=256
// ---------------------------------------------------------------------------

#define Ndim 8192
#define Fdim 512
#define Cdim 256

typedef __attribute__((ext_vector_type(16))) __bf16 v16bf;
typedef __attribute__((ext_vector_type(8)))  float  v8f;
typedef __attribute__((ext_vector_type(2)))  float  v2f;

__device__ __forceinline__ unsigned short f2bf_rne(float f) {
  unsigned int x = __builtin_bit_cast(unsigned int, f);
  unsigned int r = x + 0x7FFFu + ((x >> 16) & 1u);
  return (unsigned short)(r >> 16);
}

// ---------------------------------------------------------------------------
// Kernel 1: dis[i] = rsqrt(sum_j A[i,j] + 1 + 1e-10)   (streams A once)
// ---------------------------------------------------------------------------
__global__ __launch_bounds__(256) void gcn_deg_kernel(const float* __restrict__ A,
                                                      float* __restrict__ dis) {
  __shared__ float red[256];
  const int row = blockIdx.x;
  const float* ar = A + (size_t)row * Ndim;
  float s = 0.f;
  for (int j = threadIdx.x; j < Ndim; j += 256) s += ar[j];
  red[threadIdx.x] = s;
  __syncthreads();
  for (int off = 128; off > 0; off >>= 1) {
    if (threadIdx.x < off) red[threadIdx.x] += red[threadIdx.x + off];
    __syncthreads();
  }
  if (threadIdx.x == 0) dis[row] = rsqrtf(red[0] + 1.0f + 1e-10f);
}

// ---------------------------------------------------------------------------
// Kernel 2: Y = dis[:,None] * (X @ W)  via v_wmma_f32_16x16x4_f32 (exact fp32)
//   Yf : fp32 [N][C]        (for the +I*Y term in the epilogue)
//   Yt : bf16 [C][N]        (transposed -> contiguous WMMA B fragments)
// Block = 128 threads (4 waves). Block owns 16 rows x all 256 cols.
// Wave w owns n-tiles [4w..4w+3] (acc = 4 x v8f = 32 VGPRs).
// ---------------------------------------------------------------------------
__global__ __launch_bounds__(128) void gcn_xw_kernel(const float* __restrict__ X,
                                                     const float* __restrict__ W,
                                                     const float* __restrict__ dis,
                                                     float* __restrict__ Yf,
                                                     unsigned short* __restrict__ Yt) {
  const int lane = threadIdx.x & 31;
  const int wave = threadIdx.x >> 5;
  const int m0   = blockIdx.x * 16;
  const int r16  = lane & 15;
  const bool lo  = lane < 16;

  v8f acc[4] = {};
  const float* xrow = X + (size_t)(m0 + r16) * Fdim;

  for (int kk = 0; kk < Fdim / 4; ++kk) {
    // A fragment (16x4 f32): lane<16 -> K={k0,k0+1}, lane>=16 -> K={k0+2,k0+3}
    const int kb = kk * 4 + (lo ? 0 : 2);
    v2f af;
    {
      const float2 t = *(const float2*)(xrow + kb);
      af.x = t.x; af.y = t.y;
    }
#pragma unroll
    for (int t = 0; t < 4; ++t) {
      // B fragment (4x16 f32): col N = lane%16, same K pair as A's lane half
      const int n = wave * 64 + t * 16 + r16;
      v2f bf;
      bf.x = W[kb * Cdim + n];
      bf.y = W[(kb + 1) * Cdim + n];
      acc[t] = __builtin_amdgcn_wmma_f32_16x16x4_f32(
          false, af, false, bf, (short)0, acc[t], false, false);
    }
  }

  // Epilogue: scale by dis[m]; write fp32 Y and bf16 Y^T.
  // C/D layout: VGPR r, lanes0-15 -> M=r, lanes16-31 -> M=8+r, N=lane%16.
#pragma unroll
  for (int t = 0; t < 4; ++t) {
    const int n = wave * 64 + t * 16 + r16;
#pragma unroll
    for (int r = 0; r < 8; ++r) {
      const int m = m0 + r + (lo ? 0 : 8);
      const float v = dis[m] * acc[t][r];
      Yf[(size_t)m * Cdim + n] = v;
      Yt[(size_t)n * Ndim + m] = f2bf_rne(v);
    }
  }
}

// ---------------------------------------------------------------------------
// Kernel 3: out = relu(dis * (A @ Y + Y))  via v_wmma_f32_16x16x32_bf16.
// Block = 128 threads (4 waves), owns 16 rows x 256 cols; A is read ONCE from
// HBM: cooperatively staged fp32->bf16 into double-buffered LDS tiles (16x32),
// shared by all 4 waves. B fragments come straight from L2-resident Yt.
// ---------------------------------------------------------------------------
__global__ __launch_bounds__(128) void gcn_agg_kernel(const float* __restrict__ A,
                                                      const float* __restrict__ dis,
                                                      const float* __restrict__ Yf,
                                                      const unsigned short* __restrict__ Yt,
                                                      float* __restrict__ out) {
  // two 16x32 bf16 tiles, row stride 32 elements (64 B)
  __shared__ __align__(16) unsigned short ldsA[2][16 * 32];

  const int lane = threadIdx.x & 31;
  const int wave = threadIdx.x >> 5;
  const int m0   = blockIdx.x * 16;
  const int r16  = lane & 15;
  const bool lo  = lane < 16;

  // staging: 512 fp32 elements / 128 threads = 4 contiguous per thread
  const int e0   = threadIdx.x * 4;
  const int srow = e0 >> 5;   // 0..15
  const int skk  = e0 & 31;   // multiple of 4
  const float* agrow = A + (size_t)(m0 + srow) * Ndim;

  v8f acc[4] = {};

  // prologue: stage k-step 0
  float4 stage = *(const float4*)(agrow + skk);
  {
    const unsigned int p0 = (unsigned)f2bf_rne(stage.x) | ((unsigned)f2bf_rne(stage.y) << 16);
    const unsigned int p1 = (unsigned)f2bf_rne(stage.z) | ((unsigned)f2bf_rne(stage.w) << 16);
    *(uint2*)&ldsA[0][srow * 32 + skk] = make_uint2(p0, p1);
  }
  __syncthreads();

  const int KSTEPS = Ndim / 32;
  for (int ks = 0; ks < KSTEPS; ++ks) {
    const int k0 = ks * 32;

    // prefetch next A tile from global (overlaps with WMMA below)
    if (ks + 1 < KSTEPS)
      stage = *(const float4*)(agrow + (ks + 1) * 32 + skk);

    // A fragment (16x32 bf16) from LDS per ISA layout:
    //   lane<16 : row=lane,    K 0-7 (bytes 0-15)  + K 16-23 (bytes 32-47)
    //   lane>=16: row=lane-16, K 8-15 (bytes 16-31)+ K 24-31 (bytes 48-63)
    v16bf afrag;
    {
      const unsigned short* base = &ldsA[ks & 1][r16 * 32 + (lo ? 0 : 8)];
      union { uint4 q[2]; v16bf v; } ua;
      ua.q[0] = *(const uint4*)(base);
      ua.q[1] = *(const uint4*)(base + 16);
      afrag = ua.v;
    }

#pragma unroll
    for (int t = 0; t < 4; ++t) {
      // B fragment (32x16 bf16): col N = lane%16; lane<16 -> K 0-7 + 16-23,
      // lane>=16 -> K 8-15 + 24-31 (same as A) ; from Yt[n][k]
      const int n = wave * 64 + t * 16 + r16;
      const unsigned short* bp = Yt + (size_t)n * Ndim + k0 + (lo ? 0 : 8);
      union { uint4 q[2]; v16bf v; } ub;
      ub.q[0] = *(const uint4*)(bp);
      ub.q[1] = *(const uint4*)(bp + 16);
      acc[t] = __builtin_amdgcn_wmma_f32_16x16x32_bf16(
          false, afrag, false, ub.v, (short)0, acc[t], false, false);
    }

    // store prefetched tile into the other LDS buffer (safe: its last readers
    // synced at the end of the previous iteration)
    if (ks + 1 < KSTEPS) {
      const unsigned int p0 = (unsigned)f2bf_rne(stage.x) | ((unsigned)f2bf_rne(stage.y) << 16);
      const unsigned int p1 = (unsigned)f2bf_rne(stage.z) | ((unsigned)f2bf_rne(stage.w) << 16);
      *(uint2*)&ldsA[(ks + 1) & 1][srow * 32 + skk] = make_uint2(p0, p1);
    }
    __syncthreads();
  }

  // Epilogue: out = relu(dis[m] * (acc + Y[m,n]))
#pragma unroll
  for (int t = 0; t < 4; ++t) {
    const int n = wave * 64 + t * 16 + r16;
#pragma unroll
    for (int r = 0; r < 8; ++r) {
      const int m = m0 + r + (lo ? 0 : 8);
      const float v = dis[m] * (acc[t][r] + Yf[(size_t)m * Cdim + n]);
      out[(size_t)m * Cdim + n] = v > 0.f ? v : 0.f;
    }
  }
}

// ---------------------------------------------------------------------------
extern "C" void kernel_launch(void* const* d_in, const int* in_sizes, int n_in,
                              void* d_out, int out_size, void* d_ws, size_t ws_size,
                              hipStream_t stream) {
  const float* X = (const float*)d_in[0];   // [8192, 512]
  const float* A = (const float*)d_in[1];   // [8192, 8192]
  const float* W = (const float*)d_in[2];   // [512, 256]
  float* out = (float*)d_out;               // [8192, 256]

  char* ws = (char*)d_ws;
  float* dis          = (float*)ws;                                   // 32 KB
  float* Yf           = (float*)(ws + 65536);                         // 8 MB
  unsigned short* Yt  = (unsigned short*)(ws + 65536 +
                         (size_t)Ndim * Cdim * sizeof(float));        // 4 MB

  gcn_deg_kernel<<<Ndim, 256, 0, stream>>>(A, dis);
  gcn_xw_kernel<<<Ndim / 16, 128, 0, stream>>>(X, W, dis, Yf, Yt);
  gcn_agg_kernel<<<Ndim / 16, 128, 0, stream>>>(A, dis, Yf, Yt, out);
}
